// MyRNN_11175504904257
// MI455X (gfx1250) — matrix operand order repaired and checked by reading, not compiled, not run on hardware
//
#include <hip/hip_runtime.h>

// ---------------------------------------------------------------------------
// MyRNN on MI455X (gfx1250):
//   prep:  weights fp32 [K][N] -> bf16 transposed [N][K] (tiled transpose)
//   per T-chunk:
//     convx: x fp32 -> bf16 (streaming)
//     gemm:  bf16 WMMA; double-buffered LDS filled with
//            global_load_async_to_lds_b128 (ASYNCcnt), one barrier/K-step
//     scan:  y = y*G + H sequentially over the chunk
// ---------------------------------------------------------------------------

#define INP   512
#define OUTN  512
#define BATCH 64
#define TT    2048

typedef __attribute__((ext_vector_type(16))) __bf16 v16bf;
typedef __attribute__((ext_vector_type(8)))  float  v8f;

static __device__ __forceinline__ unsigned short f2bf(float f) {
    unsigned int u = __float_as_uint(f);
    u += 0x7FFFu + ((u >> 16) & 1u);            // round-to-nearest-even
    return (unsigned short)(u >> 16);
}
static __device__ __forceinline__ float bf2f(unsigned short s) {
    return __uint_as_float(((unsigned int)s) << 16);
}

// ---- prep: transpose+convert weights to bf16 [n][k], n in [0,1024) --------
__global__ __launch_bounds__(256)
void conv_weights_t(const float* __restrict__ g, const float* __restrict__ h,
                    unsigned short* __restrict__ wT) {
    __shared__ float tile[32][33];
    const int kb = blockIdx.x * 32;          // K base
    const int nb = blockIdx.y * 32;          // combined-N base
    const float* src = (nb < OUTN) ? g : h;
    const int noff = nb & (OUTN - 1);
    const int tid = threadIdx.x;
    const int c = tid & 31;
    #pragma unroll
    for (int i = 0; i < 4; ++i) {
        int rr = (tid >> 5) + i * 8;
        tile[rr][c] = src[(size_t)(kb + rr) * OUTN + noff + c];   // coalesced read
    }
    __syncthreads();
    #pragma unroll
    for (int i = 0; i < 4; ++i) {
        int rr = (tid >> 5) + i * 8;                              // n offset
        wT[(size_t)(nb + rr) * INP + kb + c] = f2bf(tile[c][rr]); // coalesced write
    }
}

// ---- per-chunk: x fp32 -> bf16 (8 elems / thread) -------------------------
__global__ __launch_bounds__(256)
void conv_x(const float* __restrict__ x, unsigned short* __restrict__ xbf,
            int t0) {
    size_t base = ((size_t)blockIdx.x * 256 + threadIdx.x) * 8;
    const float4 f0 = *(const float4*)(x + (size_t)t0 * BATCH * INP + base);
    const float4 f1 = *(const float4*)(x + (size_t)t0 * BATCH * INP + base + 4);
    uint4 o;
    o.x = ((unsigned int)f2bf(f0.y) << 16) | f2bf(f0.x);
    o.y = ((unsigned int)f2bf(f0.w) << 16) | f2bf(f0.z);
    o.z = ((unsigned int)f2bf(f1.y) << 16) | f2bf(f1.x);
    o.w = ((unsigned int)f2bf(f1.w) << 16) | f2bf(f1.z);
    *(uint4*)(xbf + base) = o;
}

// ---- GEMM + activations for one T-chunk -----------------------------------
// block (tl, nb): timestep t = t0+tl, combined cols [nb*128, nb*128+128)
// of N_total = 1024 (cols 0..511 -> G sigmoid, 512..1023 -> H tanh).
__global__ __launch_bounds__(256)
void rnn_gates_gemm(const unsigned short* __restrict__ xbf,
                    const unsigned short* __restrict__ wT,
                    const float* __restrict__ gb, const float* __restrict__ hb,
                    const float* __restrict__ r,  const float* __restrict__ rb,
                    unsigned short* __restrict__ gateG,
                    unsigned short* __restrict__ gateH,
                    int t0) {
    // Double-buffered tiles: A = 64x32 bf16 (4 KB), B = 128x32 bf16 (8 KB).
    __shared__ __align__(16) unsigned short Als[2 * BATCH * 32];
    __shared__ __align__(16) unsigned short Bls[2 * 128 * 32];

    const int tid  = threadIdx.x;
    const int lane = tid & 31;
    const int w    = tid >> 5;            // 8 waves
    const int tl   = blockIdx.x;
    const int t    = t0 + tl;
    const int n0   = blockIdx.y * 128;    // combined-N base
    const bool gpath = (n0 < OUTN);

    const int mtile = w & 3;              // 16 batch rows per wave
    const int nhalf = w >> 2;             // 64 cols per wave (4 n-tiles)

    v8f acc[4];
    {
        v8f z = {0.f, 0.f, 0.f, 0.f, 0.f, 0.f, 0.f, 0.f};
        acc[0] = z; acc[1] = z; acc[2] = z; acc[3] = z;
    }

    unsigned int* Als32 = (unsigned int*)Als;
    unsigned int* Bls32 = (unsigned int*)Bls;

    // Per-thread staging coordinates (constant across K loop).
    const unsigned short* xb = xbf + (size_t)tl * BATCH * INP;
    const int arow  = tid >> 2;                // 0..63
    const int akoff = (tid & 3) * 8;           // bf16 offset within 32
    const int bn0 = (tid * 2) >> 2;            // B segment 0: n
    const int bk0 = ((tid * 2) & 3) * 8;
    const int bn1 = (tid * 2 + 1) >> 2;        // B segment 1: n
    const int bk1 = ((tid * 2 + 1) & 3) * 8;
    const unsigned short* wB = wT + (size_t)n0 * INP;

    // Global byte addresses for this thread's three 16B segments at ko=0.
    unsigned long long gA  = (unsigned long long)(size_t)(xb + (size_t)arow * INP + akoff);
    unsigned long long gB0 = (unsigned long long)(size_t)(wB + (size_t)bn0 * INP + bk0);
    unsigned long long gB1 = (unsigned long long)(size_t)(wB + (size_t)bn1 * INP + bk1);
    // LDS byte addresses (low 32 bits of the shared-aperture address).
    unsigned lA  = (unsigned)(size_t)&Als[arow * 32 + akoff];
    unsigned lB0 = (unsigned)(size_t)&Bls[bn0 * 32 + bk0];
    unsigned lB1 = (unsigned)(size_t)&Bls[bn1 * 32 + bk1];

    // Issue the three async global->LDS copies for K-slice `ko` into buffer `buf`.
    auto issue = [&](int buf, int ko) {
        unsigned long long off = (unsigned long long)(ko * 2); // bytes along K
        unsigned long long a  = gA  + off;
        unsigned long long b0 = gB0 + off;
        unsigned long long b1 = gB1 + off;
        unsigned la  = lA  + buf * (BATCH * 32 * 2);
        unsigned lb0 = lB0 + buf * (128 * 32 * 2);
        unsigned lb1 = lB1 + buf * (128 * 32 * 2);
        asm volatile("global_load_async_to_lds_b128 %0, %1, off"
                     :: "v"(la), "v"(a) : "memory");
        asm volatile("global_load_async_to_lds_b128 %0, %1, off"
                     :: "v"(lb0), "v"(b0) : "memory");
        asm volatile("global_load_async_to_lds_b128 %0, %1, off"
                     :: "v"(lb1), "v"(b1) : "memory");
    };

    issue(0, 0);                               // prologue: fill buffer 0

    int buf = 0;
    for (int ko = 0; ko < INP; ko += 32, buf ^= 1) {
        asm volatile("s_wait_asynccnt 0x0" ::: "memory"); // my cur loads landed
        __syncthreads();                                  // everyone's landed
        if (ko + 32 < INP) issue(buf ^ 1, ko + 32);       // overlap next slice

        const unsigned int* Acur = Als32 + buf * (BATCH * 32 / 2);
        const unsigned int* Bcur = Bls32 + buf * (128 * 32 / 2);

        // A fragment: 16x32 bf16 (lane = M%16 + 16*half; VGPR v holds
        // K pair k0 = (v%4)*2 + half*8 + (v/4)*16).
        union { unsigned int u[8]; v16bf v; } af;
        {
            int m    = mtile * 16 + (lane & 15);
            int half = lane >> 4;
            #pragma unroll
            for (int vi = 0; vi < 8; ++vi) {
                int k0 = ((vi & 3) << 1) + half * 8 + ((vi >> 2) << 4);
                af.u[vi] = Acur[(m * 32 + k0) >> 1];
            }
        }
        // All four B fragments first, then back-to-back WMMAs.
        union { unsigned int u[8]; v16bf v; } bfr[4];
        #pragma unroll
        for (int nt = 0; nt < 4; ++nt) {
            int n    = nhalf * 64 + nt * 16 + (lane & 15);
            int half = lane >> 4;
            #pragma unroll
            for (int vi = 0; vi < 8; ++vi) {
                int k0 = ((vi & 3) << 1) + half * 8 + ((vi >> 2) << 4);   // same K order as A
                bfr[nt].u[vi] = Bcur[(n * 32 + k0) >> 1];
            }
        }
        #pragma unroll
        for (int nt = 0; nt < 4; ++nt) {
            acc[nt] = __builtin_amdgcn_wmma_f32_16x16x32_bf16(
                false, af.v, false, bfr[nt].v, (short)0, acc[nt], false, false);
        }
    }

    // Epilogue: bias + rt gate + sigmoid/tanh, store bf16 gates.
    const float tfrac = (float)t / (float)TT;
    unsigned short* dst = gpath ? gateG : gateH;
    #pragma unroll
    for (int nt = 0; nt < 4; ++nt) {
        int ncol = n0 + nhalf * 64 + nt * 16 + (lane & 15);
        int o    = gpath ? ncol : (ncol - OUTN);
        float bias = gpath ? gb[o] : hb[o];
        float rt   = 2.0f / (1.0f + __expf(-(tfrac * r[o] + rb[o])));
        #pragma unroll
        for (int vi = 0; vi < 8; ++vi) {
            int m = vi + ((lane >> 4) << 3);   // C layout: M = v + 8*(lane/16)
            int b = mtile * 16 + m;
            float pre = acc[nt][vi] + bias;
            float val;
            if (gpath) val = rt / (1.0f + __expf(-pre));                       // sigmoid
            else       val = rt * (1.0f - 2.0f / (1.0f + __expf(2.0f * pre))); // tanh
            dst[(size_t)tl * (BATCH * OUTN) + (size_t)b * OUTN + o] = f2bf(val);
        }
    }
}

// ---- sequential scan over this T-chunk ------------------------------------
__global__ __launch_bounds__(256)
void rnn_scan(const unsigned short* __restrict__ gateG,
              const unsigned short* __restrict__ gateH,
              float* __restrict__ y, int t0, int tc) {
    int idx = blockIdx.x * 256 + threadIdx.x;  // 0 .. B*OUT-1
    float acc = (t0 == 0) ? 0.0f : y[idx];
    size_t off = (size_t)idx;
    for (int tl = 0; tl < tc; ++tl) {
        float gv = bf2f(gateG[off]);
        float hv = bf2f(gateH[off]);
        acc = fmaf(acc, gv, hv);
        off += BATCH * OUTN;
    }
    y[idx] = acc;
}

// ---------------------------------------------------------------------------
extern "C" void kernel_launch(void* const* d_in, const int* in_sizes, int n_in,
                              void* d_out, int out_size, void* d_ws, size_t ws_size,
                              hipStream_t stream) {
    const float* x  = (const float*)d_in[0];
    const float* g  = (const float*)d_in[1];
    const float* gb = (const float*)d_in[2];
    const float* h  = (const float*)d_in[3];
    const float* hb = (const float*)d_in[4];
    const float* r  = (const float*)d_in[5];
    const float* rb = (const float*)d_in[6];
    float* y = (float*)d_out;

    char* ws = (char*)d_ws;
    unsigned short* wT = (unsigned short*)ws;                       // 1 MB
    const size_t wbytes = (size_t)2 * OUTN * INP * sizeof(unsigned short);
    // per-timestep ws: x_bf16 (64 KB) + G + H gates (2 * 64 KB) = 192 KB
    const size_t elems_per_t = (size_t)BATCH * INP;                 // 32768
    const size_t bytes_per_t = elems_per_t * 2 * 3;

    size_t tcap = 1;
    if (ws_size > wbytes) {
        tcap = (ws_size - wbytes) / bytes_per_t;
        if (tcap < 1) tcap = 1;
        if (tcap > TT) tcap = TT;
    }
    unsigned short* xbf = (unsigned short*)(ws + wbytes);
    unsigned short* gG  = xbf + tcap * elems_per_t;
    unsigned short* gH  = gG  + tcap * elems_per_t;

    conv_weights_t<<<dim3(INP / 32, (2 * OUTN) / 32), 256, 0, stream>>>(g, h, wT);

    for (int t0 = 0; t0 < TT; t0 += (int)tcap) {
        int tc = TT - t0;
        if (tc > (int)tcap) tc = (int)tcap;
        conv_x<<<dim3(tc * 16), 256, 0, stream>>>(x, xbf, t0);
        rnn_gates_gemm<<<dim3(tc, 8), 256, 0, stream>>>(
            xbf, wT, gb, hb, r, rb, gG, gH, t0);
        rnn_scan<<<dim3((BATCH * OUTN) / 256), 256, 0, stream>>>(gG, gH, y, t0, tc);
    }
}
